// CustomRNNmodel_45715631899366
// MI455X (gfx1250) — matrix operand order repaired and checked by reading, not compiled, not run on hardware
//
#include <hip/hip_runtime.h>
#include <cstdint>
#include <cstddef>

// ---------------- problem constants ----------------
#define BB 8
#define TT 512
#define EE 1024
#define VV 50257
#define VP 50304            // V padded to multiple of 128
#define LL 2
#define NWG_RNN 64          // E/16 n-tiles

typedef __attribute__((ext_vector_type(16))) __bf16 v16bf;
typedef __attribute__((ext_vector_type(8)))  float  v8f;
typedef __attribute__((ext_vector_type(4)))  int    v4i;

// ---------------- scalar bf16 helpers ----------------
static __device__ __forceinline__ unsigned short f2bf(float f) {
    unsigned u = __builtin_bit_cast(unsigned, f);
    unsigned r = (u + 0x7fffu + ((u >> 16) & 1u)) >> 16;   // RNE
    return (unsigned short)r;
}
static __device__ __forceinline__ float bf2f(unsigned short h) {
    unsigned u = ((unsigned)h) << 16;
    return __builtin_bit_cast(float, u);
}

// CDNA5 has hardware V_TANH_F32; use it when the builtin exists.
static __device__ __forceinline__ float fast_tanh(float x) {
#if __has_builtin(__builtin_amdgcn_tanh_f32)
    return __builtin_amdgcn_tanh_f32(x);
#elif __has_builtin(__builtin_amdgcn_tanhf)
    return __builtin_amdgcn_tanhf(x);
#else
    return tanhf(x);
#endif
}

// ---------------- async global->LDS staging (ASYNCcnt path) ----------------
#if __has_builtin(__builtin_amdgcn_global_load_async_to_lds_b128) && \
    __has_builtin(__builtin_amdgcn_s_wait_asynccnt)
#define HAVE_ASYNC_LDS 1
#else
#define HAVE_ASYNC_LDS 0
#endif

#if HAVE_ASYNC_LDS
typedef __attribute__((address_space(1))) v4i GV4;   // global int4
typedef __attribute__((address_space(3))) v4i LV4;   // LDS int4
#endif

static __device__ __forceinline__ void async_copy_b128(unsigned short* ldst,
                                                       const unsigned short* gsrc) {
#if HAVE_ASYNC_LDS
    __builtin_amdgcn_global_load_async_to_lds_b128((GV4*)gsrc, (LV4*)ldst, 0, 0);
#else
    *reinterpret_cast<uint4*>(ldst) = *reinterpret_cast<const uint4*>(gsrc);
#endif
}
static __device__ __forceinline__ void async_copy_wait() {
#if HAVE_ASYNC_LDS
    __builtin_amdgcn_s_wait_asynccnt(0);
#endif
}

// ---------------- WMMA fragment loads ----------------
// A-matrix 16x32 bf16 (ISA 7.12.2): lane L -> row m=L&15, k-offset (L>>4)*8;
// halves 0..7 at k0+koff, halves 8..15 at k0+16+koff  (two 16B contiguous chunks)
static __device__ __forceinline__ v16bf load_frag_a(const unsigned short* base, int lane, int k0) {
    int m    = lane & 15;
    int koff = (lane >> 4) * 8;
    const unsigned short* p = base + (size_t)m * EE + k0 + koff;
    union { uint4 u[2]; v16bf v; } w;
    w.u[0] = *reinterpret_cast<const uint4*>(p);
    w.u[1] = *reinterpret_cast<const uint4*>(p + 16);
    return w.v;
}
// B-matrix 32x16 bf16 for  out = x @ W^T  with W row-major [n][k]:
// lane L -> col n=L&15, k-offset 8*(L>>4); halves 0..7 at k0+koff, 8..15 at k0+16+koff
static __device__ __forceinline__ v16bf load_frag_b(const unsigned short* base, int lane, int k0) {
    int n  = lane & 15;
    int kh = (lane >> 4) * 8;
    const unsigned short* p = base + (size_t)n * EE + k0 + kh;
    union { uint4 u[2]; v16bf v; } w;
    w.u[0] = *reinterpret_cast<const uint4*>(p);
    w.u[1] = *reinterpret_cast<const uint4*>(p + 16);
    return w.v;
}
static __device__ __forceinline__ v8f wmma_bf16(v16bf a, v16bf b, v8f c) {
    return __builtin_amdgcn_wmma_f32_16x16x32_bf16(false, a, false, b, (short)0, c, false, false);
}

// ---------------- kernels ----------------

// feats[t][r][e] = bf16( W_emb[ids[r][t]][e] + W_pos[t][e] ), rows 8..15 zero-padded
__global__ void __launch_bounds__(256) embed_kernel(const int* __restrict__ ids,
                                                    const float* __restrict__ wemb,
                                                    const float* __restrict__ wpos,
                                                    unsigned short* __restrict__ feats) {
    int row = blockIdx.x;            // 0 .. T*16-1
    int t = row >> 4, r = row & 15;
    unsigned short* dst = feats + (size_t)row * EE;
    if (r < BB) {
        int id = ids[r * TT + t];
        const float* we = wemb + (size_t)id * EE;
        const float* wp = wpos + (size_t)t * EE;
        for (int e = threadIdx.x; e < EE; e += 256)
            dst[e] = f2bf(we[e] + wp[e]);
    } else {
        for (int e = threadIdx.x; e < EE; e += 256)
            dst[e] = 0;
    }
}

__global__ void __launch_bounds__(256) cvt_f32_bf16_kernel(const float* __restrict__ s,
                                                           unsigned short* __restrict__ d,
                                                           size_t n) {
    size_t i = (size_t)blockIdx.x * blockDim.x + threadIdx.x;
    size_t stride = (size_t)gridDim.x * blockDim.x;
    for (; i < n; i += stride) d[i] = f2bf(s[i]);
}

// grid-wide split barrier over global atomics (persistent-kernel sync)
static __device__ __forceinline__ void grid_barrier(unsigned* cnt, unsigned* gen, unsigned nwg) {
    __syncthreads();
    if (threadIdx.x == 0) {
        __builtin_amdgcn_fence(__ATOMIC_RELEASE, "agent");
        volatile unsigned* vgen = gen;
        unsigned g = *vgen;
        if (atomicAdd(cnt, 1u) == nwg - 1u) {
            *cnt = 0u;
            __builtin_amdgcn_fence(__ATOMIC_RELEASE, "agent");
            atomicAdd(gen, 1u);
        } else {
            while (*vgen == g) { __builtin_amdgcn_s_sleep(1); }
        }
        __builtin_amdgcn_fence(__ATOMIC_ACQUIRE, "agent");
    }
    __syncthreads();
}

// one RNN cell slice: h_new[0:8][n0:n0+16] = tanh(x@Wih^T + h@Whh^T + bias)
static __device__ __forceinline__ void rnn_cell(const unsigned short* __restrict__ x,
                                                const unsigned short* __restrict__ hp,
                                                const unsigned short* __restrict__ Wih,
                                                const unsigned short* __restrict__ Whh,
                                                float bias, int lane, int n0,
                                                unsigned short* __restrict__ hdst,
                                                unsigned short* __restrict__ odst) {
    v8f cih = {0.f,0.f,0.f,0.f,0.f,0.f,0.f,0.f};
    v8f chh = {0.f,0.f,0.f,0.f,0.f,0.f,0.f,0.f};
#pragma unroll 2
    for (int k0 = 0; k0 < EE; k0 += 32) {
        cih = wmma_bf16(load_frag_a(x,  lane, k0), load_frag_b(Wih, lane, k0), cih);
        chh = wmma_bf16(load_frag_a(hp, lane, k0), load_frag_b(Whh, lane, k0), chh);
    }
    if (lane < 16) {                 // lanes 0..15 hold rows M=0..7 (the valid batch)
        int n = n0 + lane;
#pragma unroll
        for (int r = 0; r < BB; ++r) {
            float v = fast_tanh(cih[r] + chh[r] + bias);
            unsigned short bv = f2bf(v);
            hdst[(size_t)r * EE + n] = bv;
            if (odst) odst[(size_t)r * EE + n] = bv;
        }
    }
}

// persistent RNN scan: 64 WGs x 1 wave; weights LDS-resident (128KB of the 320KB/WGP)
__global__ void __launch_bounds__(32) rnn_kernel(const unsigned short* __restrict__ feats,
                                                 const unsigned short* __restrict__ wih,
                                                 const unsigned short* __restrict__ whh,
                                                 const float* __restrict__ bih,
                                                 const float* __restrict__ bhh,
                                                 unsigned short* __restrict__ hbuf,
                                                 unsigned short* __restrict__ rnnout,
                                                 unsigned* __restrict__ sync) {
    extern __shared__ __align__(16) unsigned short lds[];   // 4 * 16 * 1024 halves = 128KB
    const int tid = threadIdx.x;
    const int n0  = blockIdx.x * 16;

    // stage W_ih/W_hh slices for both layers into LDS via async-to-LDS (ASYNCcnt)
    for (int i = tid; i < 8192; i += 32) {                  // 8192 uint4 = 128KB
        int s   = i >> 11;            // slice: 0=Wih0 1=Whh0 2=Wih1 3=Whh1
        int rem = i & 2047;
        int r   = rem >> 7;           // row in slice
        int c   = rem & 127;          // uint4 within row
        int layer = s >> 1;
        const unsigned short* wsrc = (s & 1) ? whh : wih;
        size_t off = (size_t)layer * EE * EE + (size_t)(n0 + r) * EE + (size_t)c * 8;
        async_copy_b128(lds + (size_t)i * 8, wsrc + off);
    }
    async_copy_wait();
    __syncthreads();

    const int lane = tid;
    const int nl   = lane & 15;
    const float bias0 = bih[n0 + nl]      + bhh[n0 + nl];
    const float bias1 = bih[EE + n0 + nl] + bhh[EE + n0 + nl];
    const unsigned short* Wih0 = lds;
    const unsigned short* Whh0 = lds + 16384;
    const unsigned short* Wih1 = lds + 32768;
    const unsigned short* Whh1 = lds + 49152;

    unsigned* cnt = sync;
    unsigned* gen = sync + 1;

    for (int t = 0; t < TT; ++t) {
        int p = t & 1;
        const unsigned short* x0 = feats + (size_t)t * 16 * EE;
        const unsigned short* h0p = hbuf + (size_t)(0 * 2 + (p ^ 1)) * 16 * EE;
        unsigned short*       h0n = hbuf + (size_t)(0 * 2 + p)       * 16 * EE;
        rnn_cell(x0, h0p, Wih0, Whh0, bias0, lane, n0, h0n, nullptr);
        grid_barrier(cnt, gen, NWG_RNN);

        const unsigned short* h1p = hbuf + (size_t)(2 + (p ^ 1)) * 16 * EE;
        unsigned short*       h1n = hbuf + (size_t)(2 + p)       * 16 * EE;
        rnn_cell(h0n, h1p, Wih1, Whh1, bias1, lane, n0,
                 h1n, rnnout + (size_t)t * 16 * EE);
        grid_barrier(cnt, gen, NWG_RNN);
    }
}

// LayerNorm: one block per (b,t) row; output bf16 A-matrix for the LM head
__global__ void __launch_bounds__(256) ln_kernel(const unsigned short* __restrict__ rnnout,
                                                 const float* __restrict__ g,
                                                 const float* __restrict__ bta,
                                                 unsigned short* __restrict__ xln) {
    __shared__ float red[256];
    int row = blockIdx.x;             // b*T + t
    int b = row >> 9, t = row & 511;
    const unsigned short* src = rnnout + ((size_t)t * 16 + b) * EE;
    int tid = threadIdx.x;

    float s = 0.f, s2 = 0.f;
    for (int e = tid; e < EE; e += 256) {
        float v = bf2f(src[e]);
        s += v; s2 += v * v;
    }
    red[tid] = s; __syncthreads();
    for (int o = 128; o > 0; o >>= 1) { if (tid < o) red[tid] += red[tid + o]; __syncthreads(); }
    float mu = red[0] * (1.0f / EE);
    __syncthreads();
    red[tid] = s2; __syncthreads();
    for (int o = 128; o > 0; o >>= 1) { if (tid < o) red[tid] += red[tid + o]; __syncthreads(); }
    float var = red[0] * (1.0f / EE) - mu * mu;
    float rstd = rsqrtf(var + 1e-5f);

    unsigned short* dst = xln + (size_t)row * EE;
    for (int e = tid; e < EE; e += 256) {
        float v = bf2f(src[e]);
        dst[e] = f2bf((v - mu) * rstd * g[e] + bta[e]);
    }
}

// LM head: out[4096, V] = Xln[4096,1024] @ Wemb^T  (bf16 WMMA, f32 accumulate)
// 64x128 tile per 128-thread WG; A-tile staged in LDS (128KB); 8x A-fragment reuse
__global__ void __launch_bounds__(128) lmhead_kernel(const unsigned short* __restrict__ xln,
                                                     const unsigned short* __restrict__ wemb,
                                                     float* __restrict__ out) {
    extern __shared__ __align__(16) unsigned short lds[];   // 64 * 1024 halves = 128KB
    const int n0 = blockIdx.x * 128;
    const int m0 = blockIdx.y * 64;

    const unsigned short* asrc = xln + (size_t)m0 * EE;
    for (int i = threadIdx.x; i < 8192; i += 128)
        async_copy_b128(lds + (size_t)i * 8, asrc + (size_t)i * 8);
    async_copy_wait();
    __syncthreads();

    const int wave = threadIdx.x >> 5;
    const int lane = threadIdx.x & 31;
    const unsigned short* A = lds + (size_t)wave * 16 * EE;

    const v8f vzero = {0.f,0.f,0.f,0.f,0.f,0.f,0.f,0.f};
    v8f c[8];
    const unsigned short* Bp[8];
#pragma unroll
    for (int j = 0; j < 8; ++j) {
        c[j]  = vzero;
        Bp[j] = wemb + (size_t)(n0 + j * 16) * EE;
    }
    const size_t lrow = (size_t)(lane & 15) * EE;

#pragma unroll 2
    for (int k0 = 0; k0 < EE; k0 += 32) {
        if (k0 < EE - 256) {   // pull upcoming chunks of the L2-resident Wemb rows
            __builtin_prefetch((const void*)(Bp[0] + lrow + k0 + 256), 0, 1);
            __builtin_prefetch((const void*)(Bp[4] + lrow + k0 + 256), 0, 1);
        }
        v16bf a = load_frag_a(A, lane, k0);
#pragma unroll
        for (int j = 0; j < 8; ++j)
            c[j] = wmma_bf16(a, load_frag_b(Bp[j], lane, k0), c[j]);
    }

    const int nl = lane & 15;
    const int rb = (lane >> 4) * 8;
#pragma unroll
    for (int j = 0; j < 8; ++j) {
        int n = n0 + j * 16 + nl;
        if (n < VV) {
#pragma unroll
            for (int r = 0; r < 8; ++r) {
                int m = m0 + wave * 16 + rb + r;
                out[(size_t)m * VV + n] = c[j][r];
            }
        }
    }
}

// ---------------- host launcher ----------------
extern "C" void kernel_launch(void* const* d_in, const int* in_sizes, int n_in,
                              void* d_out, int out_size, void* d_ws, size_t ws_size,
                              hipStream_t stream) {
    (void)in_sizes; (void)n_in; (void)out_size; (void)ws_size;

    const int*   ids   = (const int*)  d_in[0];
    const float* W_emb = (const float*)d_in[1];
    const float* W_pos = (const float*)d_in[2];
    const float* ln_g  = (const float*)d_in[3];
    const float* ln_b  = (const float*)d_in[4];
    const float* W_ih  = (const float*)d_in[5];
    const float* W_hh  = (const float*)d_in[6];
    const float* b_ih  = (const float*)d_in[7];
    const float* b_hh  = (const float*)d_in[8];
    float* out = (float*)d_out;

    // workspace layout (bytes)
    char* ws = (char*)d_ws;
    size_t off = 0;
    unsigned* sync = (unsigned*)(ws + off);               off += 256;
    unsigned short* feats  = (unsigned short*)(ws + off); off += (size_t)TT * 16 * EE * 2;   // 16 MB
    unsigned short* hbuf   = (unsigned short*)(ws + off); off += (size_t)4 * 16 * EE * 2;    // 128 KB
    unsigned short* wihbf  = (unsigned short*)(ws + off); off += (size_t)LL * EE * EE * 2;   // 4 MB
    unsigned short* whhbf  = (unsigned short*)(ws + off); off += (size_t)LL * EE * EE * 2;   // 4 MB
    unsigned short* rnnout = (unsigned short*)(ws + off); off += (size_t)TT * 16 * EE * 2;   // 16 MB
    unsigned short* xln    = (unsigned short*)(ws + off); off += (size_t)BB * TT * EE * 2;   // 8 MB
    unsigned short* wembbf = (unsigned short*)(ws + off); off += (size_t)VP * EE * 2;        // 103 MB

    // deterministic init each call (harness does not re-poison between replays)
    (void)hipMemsetAsync(sync, 0, 256, stream);
    (void)hipMemsetAsync(hbuf, 0, (size_t)4 * 16 * EE * 2, stream);
    (void)hipMemsetAsync(wembbf + (size_t)VV * EE, 0, (size_t)(VP - VV) * EE * 2, stream);

    embed_kernel<<<TT * 16, 256, 0, stream>>>(ids, W_emb, W_pos, feats);
    cvt_f32_bf16_kernel<<<2048, 256, 0, stream>>>(W_ih,  wihbf,  (size_t)LL * EE * EE);
    cvt_f32_bf16_kernel<<<2048, 256, 0, stream>>>(W_hh,  whhbf,  (size_t)LL * EE * EE);
    cvt_f32_bf16_kernel<<<4096, 256, 0, stream>>>(W_emb, wembbf, (size_t)VV * EE);

    rnn_kernel<<<NWG_RNN, 32, 131072, stream>>>(feats, wihbf, whhbf, b_ih, b_hh,
                                                hbuf, rnnout, sync);

    ln_kernel<<<BB * TT, 256, 0, stream>>>(rnnout, ln_g, ln_b, xln);

    lmhead_kernel<<<dim3(VP / 128, (BB * TT) / 64), 128, 131072, stream>>>(xln, wembbf, out);
}
